// Encoder_84413287236054
// MI455X (gfx1250) — matrix operand order repaired and checked by reading, not compiled, not run on hardware
//
#include <hip/hip_runtime.h>
#include <math.h>

// ---------------------------------------------------------------------------
// Types
// ---------------------------------------------------------------------------
typedef __bf16 bf16;
typedef bf16  v16bf __attribute__((ext_vector_type(16)));
typedef float v8f   __attribute__((ext_vector_type(8)));

#define ENC_DIM 1024
#define SEQ     2048
#define NTOK    4096   // 2 * 2048
#define DFF     4096
#define NHEAD   16
#define HD      64

// ---------------------------------------------------------------------------
// Tiled transpose + fp32->bf16 convert: in[K][N] f32 -> out[N][K] bf16.
// ---------------------------------------------------------------------------
__global__ __launch_bounds__(256)
void cvt_transpose_bf16(const float* __restrict__ in, bf16* __restrict__ out,
                        int K, int N) {
    __shared__ bf16 tile[32][33];
    const int tx = threadIdx.x & 31;
    const int ty = threadIdx.x >> 5;          // 0..7
    const int n0 = blockIdx.x * 32;
    const int k0 = blockIdx.y * 32;
    #pragma unroll
    for (int i = 0; i < 4; ++i)
        tile[ty * 4 + i][tx] = (bf16)in[(size_t)(k0 + ty * 4 + i) * N + n0 + tx];
    __syncthreads();
    #pragma unroll
    for (int i = 0; i < 4; ++i)
        out[(size_t)(n0 + ty * 4 + i) * K + k0 + tx] = tile[tx][ty * 4 + i];
}

// ---------------------------------------------------------------------------
// LayerNorm over D=1024, fp32 in -> bf16 out. One row per 256-thread block.
// ---------------------------------------------------------------------------
__global__ __launch_bounds__(256) void layernorm_bf16(const float* __restrict__ x,
                                                      const float* __restrict__ g,
                                                      const float* __restrict__ b,
                                                      bf16* __restrict__ out) {
    __shared__ float sred[18];
    const int row = blockIdx.x;
    const int t = threadIdx.x;
    const float* xr = x + (size_t)row * ENC_DIM;
    float4 v = ((const float4*)xr)[t];
    float s  = v.x + v.y + v.z + v.w;
    float ss = v.x * v.x + v.y * v.y + v.z * v.z + v.w * v.w;
    #pragma unroll
    for (int off = 16; off > 0; off >>= 1) {
        s  += __shfl_down(s,  off, 32);
        ss += __shfl_down(ss, off, 32);
    }
    if ((t & 31) == 0) {
        sred[2 + (t >> 5) * 2 + 0] = s;
        sred[2 + (t >> 5) * 2 + 1] = ss;
    }
    __syncthreads();
    if (t == 0) {
        float S = 0.f, SS = 0.f;
        #pragma unroll
        for (int i = 0; i < 8; ++i) { S += sred[2 + 2 * i]; SS += sred[3 + 2 * i]; }
        float mu  = S * (1.0f / ENC_DIM);
        float var = SS * (1.0f / ENC_DIM) - mu * mu;
        sred[0] = mu;
        sred[1] = rsqrtf(var + 1e-5f);
    }
    __syncthreads();
    float mu = sred[0], rs = sred[1];
    float4 gv = ((const float4*)g)[t];
    float4 bv = ((const float4*)b)[t];
    bf16* o = out + (size_t)row * ENC_DIM + t * 4;
    o[0] = (bf16)((v.x - mu) * rs * gv.x + bv.x);
    o[1] = (bf16)((v.y - mu) * rs * gv.y + bv.y);
    o[2] = (bf16)((v.z - mu) * rs * gv.z + bv.z);
    o[3] = (bf16)((v.w - mu) * rs * gv.w + bv.w);
}

// ---------------------------------------------------------------------------
// bf16 GEMM: C[M,N] = A[M,K] @ B[K,N] + bias, B supplied TRANSPOSED
// (Bt[N][K], N-major). Block tile 128x256, BK=32, 256 threads (8 waves),
// wave tile 64x64 (16 v_wmma per wave per k-step). Double-buffered LDS:
// next tile's global loads issued before compute, one barrier per k-step.
// MODE 0: bf16 store. MODE 1: GELU(exact)->bf16. MODE 2: f32 + residual.
// MODE 3: bf16 store transposed head-split [b,h,hd,s] (for V).
// Requires M%128==0, N%256==0, K%32==0.
// ---------------------------------------------------------------------------
template <int MODE>
__global__ __launch_bounds__(256)
void gemm_bf16(const bf16* __restrict__ A, const bf16* __restrict__ Bt,
               const float* __restrict__ bias, const float* __restrict__ res,
               bf16* __restrict__ outb, float* __restrict__ outf,
               int M, int N, int K) {
    __shared__ __align__(16) bf16 As[2][128][32];
    __shared__ __align__(16) bf16 Bst[2][256][32];
    const int t    = threadIdx.x;
    const int L    = t & 31;
    const int w    = t >> 5;        // 0..7
    const int wm   = w & 1;         // 64-row block within 128
    const int wn   = w >> 1;        // 64-col block within 256 (0..3)
    const int m0   = blockIdx.y * 128;
    const int n0   = blockIdx.x * 256;
    const int lcol = L & 15;
    const int half8  = (L >= 16) ? 8 : 0;   // A-frag / C-row offset
    const int half16 = (L >= 16) ? 16 : 0;  // unused (B-frag uses half8)

    const int arow = t >> 1, ach = t & 1;
    const bf16* aptr = A + (size_t)(m0 + arow) * K + ach * 16;
    const bf16* bptr = Bt + (size_t)(n0 + t) * K;

    v8f acc[4][4];
    #pragma unroll
    for (int mt = 0; mt < 4; ++mt)
        #pragma unroll
        for (int nt = 0; nt < 4; ++nt)
            #pragma unroll
            for (int e = 0; e < 8; ++e) acc[mt][nt][e] = 0.0f;

    // prologue: stage tile 0 into buffer 0
    {
        const uint4* as = (const uint4*)aptr;
        uint4* ad = (uint4*)&As[0][arow][ach * 16];
        ad[0] = as[0]; ad[1] = as[1];
        const uint4* bs = (const uint4*)bptr;
        uint4* bd = (uint4*)&Bst[0][t][0];
        bd[0] = bs[0]; bd[1] = bs[1]; bd[2] = bs[2]; bd[3] = bs[3];
    }
    __syncthreads();

    const int nk = K >> 5;
    for (int kt = 0; kt < nk; ++kt) {
        const int cur = kt & 1;
        const int nxt = cur ^ 1;
        // issue next tile's global loads early (latency hidden by WMMAs)
        uint4 ar[2], br[4];
        if (kt + 1 < nk) {
            const uint4* as = (const uint4*)(aptr + (size_t)(kt + 1) * 32);
            ar[0] = as[0]; ar[1] = as[1];
            const uint4* bs = (const uint4*)(bptr + (size_t)(kt + 1) * 32);
            br[0] = bs[0]; br[1] = bs[1]; br[2] = bs[2]; br[3] = bs[3];
        }

        v16bf afr[4], bfr[4];
        #pragma unroll
        for (int mt = 0; mt < 4; ++mt) {
            const bf16* p = &As[cur][wm * 64 + mt * 16 + lcol][half8];
            #pragma unroll
            for (int e = 0; e < 8; ++e) { afr[mt][e] = p[e]; afr[mt][8 + e] = p[16 + e]; }
        }
        #pragma unroll
        for (int nt = 0; nt < 4; ++nt) {
            const bf16* p = &Bst[cur][wn * 64 + nt * 16 + lcol][half8];
            #pragma unroll
            for (int e = 0; e < 8; ++e) { bfr[nt][e] = p[e]; bfr[nt][8 + e] = p[16 + e]; }
        }
        #pragma unroll
        for (int mt = 0; mt < 4; ++mt)
            #pragma unroll
            for (int nt = 0; nt < 4; ++nt)
                acc[mt][nt] = __builtin_amdgcn_wmma_f32_16x16x32_bf16(
                    false, afr[mt], false, bfr[nt], (short)0, acc[mt][nt], false, false);

        if (kt + 1 < nk) {
            uint4* ad = (uint4*)&As[nxt][arow][ach * 16];
            ad[0] = ar[0]; ad[1] = ar[1];
            uint4* bd = (uint4*)&Bst[nxt][t][0];
            bd[0] = br[0]; bd[1] = br[1]; bd[2] = br[2]; bd[3] = br[3];
        }
        __syncthreads();
    }

    // epilogue
    #pragma unroll
    for (int mt = 0; mt < 4; ++mt)
        #pragma unroll
        for (int nt = 0; nt < 4; ++nt) {
            int col = n0 + wn * 64 + nt * 16 + lcol;
            float bv = bias[col];
            if constexpr (MODE == 3) {
                // V: store transposed head-split [b,h,hd,s]; 8 consecutive s
                // values per lane -> one 16-byte store.
                int rowbase = m0 + wm * 64 + mt * 16 + half8;
                int bb = rowbase >> 11;          // batch
                int s0 = rowbase & (SEQ - 1);    // sequence pos of r=0
                int head = col >> 6, hd = col & (HD - 1);
                alignas(16) bf16 tmp[8];
                #pragma unroll
                for (int r = 0; r < 8; ++r) tmp[r] = (bf16)(acc[mt][nt][r] + bv);
                size_t idx = ((size_t)(bb * NHEAD + head) * HD + hd) * SEQ + s0;
                *(uint4*)&outb[idx] = *(const uint4*)tmp;
            } else {
                #pragma unroll
                for (int r = 0; r < 8; ++r) {
                    int row = m0 + wm * 64 + mt * 16 + r + half8;
                    size_t idx = (size_t)row * N + col;
                    float v = acc[mt][nt][r] + bv;
                    if constexpr (MODE == 0) {
                        outb[idx] = (bf16)v;
                    } else if constexpr (MODE == 1) {
                        float gv = 0.5f * v * (1.0f + erff(v * 0.70710678118654752f));
                        outb[idx] = (bf16)gv;
                    } else {
                        outf[idx] = v + res[idx];
                    }
                }
            }
        }
}

// ---------------------------------------------------------------------------
// Flash attention. Q/K token-major [4096,1024] bf16 (heads interleaved);
// V transposed head-split [b,h,hd,s] bf16 so PV B-fragments are contiguous.
// Block = (qtile, b*H+h); 128 threads, 4 waves x 16 q-rows; 64-key tiles.
// Double-buffered K/V staging; next tile's global loads issued one tile early.
// ---------------------------------------------------------------------------
__global__ __launch_bounds__(128)
void attention_bf16(const bf16* __restrict__ Q, const bf16* __restrict__ Kg,
                    const bf16* __restrict__ Vt, bf16* __restrict__ O) {
    __shared__ __align__(16) bf16 Ks[2][64][64];     // [key row][hd]
    __shared__ __align__(16) bf16 Vts[2][64][64];    // [hd][key row]
    __shared__ __align__(16) bf16 Ps[4][16][64];
    const int t    = threadIdx.x;
    const int L    = t & 31;
    const int w    = t >> 5;
    const int lcol = L & 15;
    const int half8  = (L >= 16) ? 8 : 0;
    const int half16 = (L >= 16) ? 16 : 0;
    const int bh = blockIdx.y;              // 0..31 == b*NHEAD + h
    const int b = bh >> 4, h = bh & 15;
    const int tok0 = b * SEQ;
    const int q0 = blockIdx.x * 64 + w * 16;  // within sequence

    const int srow = t >> 1, shc = t & 1;
    const bf16* kptr = Kg + (size_t)(tok0 + srow) * ENC_DIM + h * HD + shc * 32;
    const bf16* vptr = Vt + ((size_t)bh * HD + srow) * SEQ + shc * 32;

    // Q fragments (A-layout, 16x32 per k-step), pre-scaled by 1/8
    v16bf qf[2];
    {
        const bf16* qrow = Q + (size_t)(tok0 + q0 + lcol) * ENC_DIM + h * HD;
        #pragma unroll
        for (int ks = 0; ks < 2; ++ks) {
            const bf16* p = qrow + ks * 32 + half8;
            #pragma unroll
            for (int e = 0; e < 8; ++e) {
                qf[ks][e]     = (bf16)((float)p[e]      * 0.125f);
                qf[ks][8 + e] = (bf16)((float)p[16 + e] * 0.125f);
            }
        }
    }

    v8f oacc[4];
    float m[8], l[8];
    #pragma unroll
    for (int nt = 0; nt < 4; ++nt)
        #pragma unroll
        for (int e = 0; e < 8; ++e) oacc[nt][e] = 0.0f;
    #pragma unroll
    for (int r = 0; r < 8; ++r) { m[r] = -1e30f; l[r] = 0.0f; }

    // prologue: issue global loads for tile 0
    uint4 kr[4], vr[4];
    {
        const uint4* ks_ = (const uint4*)kptr;
        kr[0] = ks_[0]; kr[1] = ks_[1]; kr[2] = ks_[2]; kr[3] = ks_[3];
        const uint4* vs_ = (const uint4*)vptr;
        vr[0] = vs_[0]; vr[1] = vs_[1]; vr[2] = vs_[2]; vr[3] = vs_[3];
    }

    for (int kt = 0; kt < SEQ / 64; ++kt) {
        const int cur = kt & 1;
        // store staged tile, then issue next tile's loads
        {
            uint4* kd = (uint4*)&Ks[cur][srow][shc * 32];
            kd[0] = kr[0]; kd[1] = kr[1]; kd[2] = kr[2]; kd[3] = kr[3];
            uint4* vd = (uint4*)&Vts[cur][srow][shc * 32];
            vd[0] = vr[0]; vd[1] = vr[1]; vd[2] = vr[2]; vd[3] = vr[3];
        }
        if (kt + 1 < SEQ / 64) {
            const uint4* ks_ = (const uint4*)(kptr + (size_t)(kt + 1) * 64 * ENC_DIM);
            kr[0] = ks_[0]; kr[1] = ks_[1]; kr[2] = ks_[2]; kr[3] = ks_[3];
            const uint4* vs_ = (const uint4*)(vptr + (size_t)(kt + 1) * 64);
            vr[0] = vs_[0]; vr[1] = vs_[1]; vr[2] = vs_[2]; vr[3] = vs_[3];
        }
        __syncthreads();

        // S = (Q * 1/8) @ K^T : 16 x 64 per wave
        v8f sacc[4];
        #pragma unroll
        for (int nt = 0; nt < 4; ++nt) {
            #pragma unroll
            for (int e = 0; e < 8; ++e) sacc[nt][e] = 0.0f;
            #pragma unroll
            for (int ks = 0; ks < 2; ++ks) {
                v16bf bfr;
                const bf16* p = &Ks[cur][nt * 16 + lcol][ks * 32 + half8];
                #pragma unroll
                for (int e = 0; e < 8; ++e) { bfr[e] = p[e]; bfr[8 + e] = p[16 + e]; }
                sacc[nt] = __builtin_amdgcn_wmma_f32_16x16x32_bf16(
                    false, qf[ks], false, bfr, (short)0, sacc[nt], false, false);
            }
        }

        // online softmax (rows live in 16-lane halves)
        float as[8];
        #pragma unroll
        for (int r = 0; r < 8; ++r) {
            float rv = fmaxf(fmaxf(sacc[0][r], sacc[1][r]), fmaxf(sacc[2][r], sacc[3][r]));
            #pragma unroll
            for (int o = 1; o < 16; o <<= 1) rv = fmaxf(rv, __shfl_xor(rv, o, 32));
            float nm = fmaxf(m[r], rv);
            as[r] = __expf(m[r] - nm);
            m[r] = nm;
        }
        #pragma unroll
        for (int r = 0; r < 8; ++r) {
            float s = 0.0f;
            #pragma unroll
            for (int nt = 0; nt < 4; ++nt) {
                float p = __expf(sacc[nt][r] - m[r]);
                sacc[nt][r] = p;
                s += p;
            }
            #pragma unroll
            for (int o = 1; o < 16; o <<= 1) s += __shfl_xor(s, o, 32);
            l[r] = l[r] * as[r] + s;
            #pragma unroll
            for (int nt = 0; nt < 4; ++nt) oacc[nt][r] *= as[r];
        }

        // write P (C-layout) to LDS as row-major 16x64 bf16 for A-frag reload
        #pragma unroll
        for (int nt = 0; nt < 4; ++nt)
            #pragma unroll
            for (int r = 0; r < 8; ++r)
                Ps[w][r + half8][nt * 16 + lcol] = (bf16)sacc[nt][r];
        __syncthreads();

        // O += P @ V  (K dim = 64 key rows -> 2 k-steps); V frags contiguous
        #pragma unroll
        for (int ks2 = 0; ks2 < 2; ++ks2) {
            v16bf afr;
            const bf16* p = &Ps[w][lcol][ks2 * 32 + half8];
            #pragma unroll
            for (int e = 0; e < 8; ++e) { afr[e] = p[e]; afr[8 + e] = p[16 + e]; }
            #pragma unroll
            for (int nt = 0; nt < 4; ++nt) {
                v16bf bfr;
                const bf16* pv = &Vts[cur][nt * 16 + lcol][ks2 * 32 + half8];
                #pragma unroll
                for (int e = 0; e < 8; ++e) { bfr[e] = pv[e]; bfr[8 + e] = pv[16 + e]; }
                oacc[nt] = __builtin_amdgcn_wmma_f32_16x16x32_bf16(
                    false, afr, false, bfr, (short)0, oacc[nt], false, false);
            }
        }
    }

    // normalize and store (token-major, heads interleaved)
    #pragma unroll
    for (int nt = 0; nt < 4; ++nt)
        #pragma unroll
        for (int r = 0; r < 8; ++r) {
            float v = oacc[nt][r] / l[r];
            int row = tok0 + q0 + r + half8;
            O[(size_t)row * ENC_DIM + h * HD + nt * 16 + lcol] = (bf16)v;
        }
}

// ---------------------------------------------------------------------------
// Orchestration
// ---------------------------------------------------------------------------
extern "C" void kernel_launch(void* const* d_in, const int* in_sizes, int n_in,
                              void* d_out, int out_size, void* d_ws, size_t ws_size,
                              hipStream_t stream) {
    const float* x     = (const float*)d_in[0];
    const float* ln1_g = (const float*)d_in[1];
    const float* ln1_b = (const float*)d_in[2];
    const float* wq    = (const float*)d_in[3];
    const float* bq    = (const float*)d_in[4];
    const float* wk    = (const float*)d_in[5];
    const float* bk    = (const float*)d_in[6];
    const float* wv    = (const float*)d_in[7];
    const float* bv    = (const float*)d_in[8];
    const float* wo    = (const float*)d_in[9];
    const float* bo    = (const float*)d_in[10];
    const float* w1    = (const float*)d_in[11];
    const float* b1    = (const float*)d_in[12];
    const float* w2    = (const float*)d_in[13];
    const float* b2    = (const float*)d_in[14];
    const float* ln2_g = (const float*)d_in[15];
    const float* ln2_b = (const float*)d_in[16];

    char* ws = (char*)d_ws;
    size_t off = 0;
    auto alloc = [&](size_t bytes) -> void* {
        void* p = ws + off;
        off += (bytes + 255) & ~(size_t)255;
        return p;
    };
    const size_t DD  = (size_t)ENC_DIM * ENC_DIM;     // 1M
    const size_t DF  = (size_t)ENC_DIM * DFF;         // 4M
    const size_t TD  = (size_t)NTOK * ENC_DIM;        // 4M
    const size_t TF  = (size_t)NTOK * DFF;            // 16M

    bf16* wq_t  = (bf16*)alloc(DD * 2);   // [N][K] transposed bf16 weights
    bf16* wk_t  = (bf16*)alloc(DD * 2);
    bf16* wv_t  = (bf16*)alloc(DD * 2);
    bf16* wo_t  = (bf16*)alloc(DD * 2);
    bf16* w1_t  = (bf16*)alloc(DF * 2);
    bf16* w2_t  = (bf16*)alloc(DF * 2);
    bf16* h_b   = (bf16*)alloc(TD * 2);
    bf16* q_b   = (bf16*)alloc(TD * 2);
    bf16* k_b   = (bf16*)alloc(TD * 2);
    bf16* vt_b  = (bf16*)alloc(TD * 2);   // V transposed head-split [b,h,hd,s]
    bf16* at_b  = (bf16*)alloc(TD * 2);
    float* x1   = (float*)alloc(TD * 4);
    bf16* h2_b  = (bf16*)alloc(TD * 2);
    bf16* ff_b  = (bf16*)alloc(TF * 2);
    (void)ws_size; (void)in_sizes; (void)n_in; (void)out_size;

    // 1. weights -> bf16, transposed to [N][K]
    dim3 gtDD(ENC_DIM / 32, ENC_DIM / 32);
    cvt_transpose_bf16<<<gtDD, 256, 0, stream>>>(wq, wq_t, ENC_DIM, ENC_DIM);
    cvt_transpose_bf16<<<gtDD, 256, 0, stream>>>(wk, wk_t, ENC_DIM, ENC_DIM);
    cvt_transpose_bf16<<<gtDD, 256, 0, stream>>>(wv, wv_t, ENC_DIM, ENC_DIM);
    cvt_transpose_bf16<<<gtDD, 256, 0, stream>>>(wo, wo_t, ENC_DIM, ENC_DIM);
    dim3 gtW1(DFF / 32, ENC_DIM / 32);
    cvt_transpose_bf16<<<gtW1, 256, 0, stream>>>(w1, w1_t, ENC_DIM, DFF);
    dim3 gtW2(ENC_DIM / 32, DFF / 32);
    cvt_transpose_bf16<<<gtW2, 256, 0, stream>>>(w2, w2_t, DFF, ENC_DIM);

    // 2. LN1
    layernorm_bf16<<<NTOK, 256, 0, stream>>>(x, ln1_g, ln1_b, h_b);

    // 3. Q,K,V projections (M=4096, N=1024, K=1024)
    dim3 gqkv(ENC_DIM / 256, NTOK / 128);
    gemm_bf16<0><<<gqkv, 256, 0, stream>>>(h_b, wq_t, bq, nullptr, q_b, nullptr, NTOK, ENC_DIM, ENC_DIM);
    gemm_bf16<0><<<gqkv, 256, 0, stream>>>(h_b, wk_t, bk, nullptr, k_b, nullptr, NTOK, ENC_DIM, ENC_DIM);
    gemm_bf16<3><<<gqkv, 256, 0, stream>>>(h_b, wv_t, bv, nullptr, vt_b, nullptr, NTOK, ENC_DIM, ENC_DIM);

    // 4. attention
    dim3 gat(SEQ / 64, 2 * NHEAD);
    attention_bf16<<<gat, 128, 0, stream>>>(q_b, k_b, vt_b, at_b);

    // 5. x1 = attn @ wo + bo + x   (f32)
    gemm_bf16<2><<<gqkv, 256, 0, stream>>>(at_b, wo_t, bo, x, nullptr, x1, NTOK, ENC_DIM, ENC_DIM);

    // 6. LN2
    layernorm_bf16<<<NTOK, 256, 0, stream>>>(x1, ln2_g, ln2_b, h2_b);

    // 7. ffh = gelu(h2 @ w1 + b1)  (M=4096, N=4096, K=1024)
    dim3 gff1(DFF / 256, NTOK / 128);
    gemm_bf16<1><<<gff1, 256, 0, stream>>>(h2_b, w1_t, b1, nullptr, ff_b, nullptr, NTOK, DFF, ENC_DIM);

    // 8. out = ffh @ w2 + b2 + x1  (M=4096, N=1024, K=4096) -> d_out f32
    gemm_bf16<2><<<gqkv, 256, 0, stream>>>(ff_b, w2_t, b2, x1, nullptr, (float*)d_out, NTOK, ENC_DIM, DFF);
}
